// NeuralODE_39032662786481
// MI455X (gfx1250) — compile-verified, matrix operand order checked by reading, not run on hardware
//
#include <hip/hip_runtime.h>

// Neural ODE fused kernel for gfx1250 (MI455X, wave32, WMMA).
// One wave per 16-row batch tile; 256 blocks x 32 threads.
// All matmuls via v_wmma_f32_16x16x32_f16; h kept in f32 D-layout accumulators.

typedef __attribute__((ext_vector_type(16))) _Float16 v16h;
typedef __attribute__((ext_vector_type(8)))  float    v8f;

#define T_STEPS 1000
#define BATCH   4096
#define HIDD    32
#define HHD     15
#define OUTD    8
#define INITD   16

// K index held by element j of a v16h A/B operand, per CDNA5 16-bit layout:
// lanes 0-15 hold K {0..7, 16..23}; lanes 16-31 hold K {8..15, 24..31}.
static __device__ __forceinline__ int kmap(int j, int half) {
    return ((j < 8) ? j : (j + 8)) + half * 8;
}

// Build a B operand (K x N, row-major, leading dim ldn) zero-padded to 32x16.
static __device__ __forceinline__ v16h load_b(const float* __restrict__ W,
                                              int K, int N, int ldn,
                                              int n, int half) {
    v16h b;
#pragma unroll
    for (int j = 0; j < 16; ++j) {
        int k = kmap(j, half);
        float v = (k < K && n < N) ? W[k * ldn + n] : 0.0f;
        b[j] = (_Float16)v;
    }
    return b;
}

// Build an A operand (16 x kvalid, zero-padded to 16x32) from a row-major LDS tile.
static __device__ __forceinline__ v16h load_a(const float* __restrict__ tile,
                                              int ld, int kvalid,
                                              int row, int half) {
    v16h a;
#pragma unroll
    for (int j = 0; j < 16; ++j) {
        int k = kmap(j, half);
        float v = (k < kvalid) ? tile[row * ld + k] : 0.0f;
        a[j] = (_Float16)v;
    }
    return a;
}

__global__ __launch_bounds__(32)
void node_fused_kernel(const float* __restrict__ times,
                       const float* __restrict__ initial,
                       const float* __restrict__ Wi,  const float* __restrict__ bi,
                       const float* __restrict__ Wf0, const float* __restrict__ bf0,
                       const float* __restrict__ Wf1, const float* __restrict__ bf1,
                       const float* __restrict__ Wf2, const float* __restrict__ bf2,
                       const float* __restrict__ Wf3, const float* __restrict__ bf3,
                       const float* __restrict__ Wl,  const float* __restrict__ bl,
                       float* __restrict__ out)
{
    __shared__ float ldsH[16 * 32];  // h staging tile (f32, row-major)
    __shared__ float ldsZ[16 * 16];  // tanh-activation staging tile

    const int lane = threadIdx.x & 31;
    const int half = lane >> 4;   // lane group
    const int col  = lane & 15;   // N (D layout) / row (A layout)
    const int mofs = 8 * half;    // D-layout row offset of vgpr 0
    const int rowbase = blockIdx.x * 16;

    // ---- Resident weights as WMMA B operands (zero-padded to 32x16) ----
    const v16h BiA = load_b(Wi,  INITD, HIDD, HIDD, col,      half);
    const v16h BiB = load_b(Wi,  INITD, HIDD, HIDD, col + 16, half);
    const v16h B0  = load_b(Wf0, HIDD,  HHD,  HHD,  col,      half);
    const v16h B1  = load_b(Wf1, HHD,   HHD,  HHD,  col,      half);
    const v16h B2  = load_b(Wf2, HHD,   HHD,  HHD,  col,      half);
    const v16h B3A = load_b(Wf3, HHD,   HIDD, HIDD, col,      half);
    const v16h B3B = load_b(Wf3, HHD,   HIDD, HIDD, col + 16, half);
    const v16h BL  = load_b(Wl,  HIDD,  OUTD, OUTD, col,      half);

    // Per-lane bias values (D layout: bias indexed by output column)
    const float biA = bi[col], biB = bi[col + 16];
    const float b0v = (col < HHD) ? bf0[col] : 0.0f;
    const float b1v = (col < HHD) ? bf1[col] : 0.0f;
    const float b2v = (col < HHD) ? bf2[col] : 0.0f;
    const float b3A = bf3[col],  b3B = bf3[col + 16];
    const float blv = (col < OUTD) ? bl[col] : 0.0f;

    const v8f zero8 = {};

    // ---- h0 = initial @ Wi + bi : A operand read directly from global ----
    v16h a0;
#pragma unroll
    for (int j = 0; j < 16; ++j) {
        int k = kmap(j, half);
        float v = (k < INITD) ? initial[(rowbase + col) * INITD + k] : 0.0f;
        a0[j] = (_Float16)v;
    }
    v8f hA = __builtin_amdgcn_wmma_f32_16x16x32_f16(false, a0, false, BiA,
                                                    (short)0, zero8, false, false);
    v8f hB = __builtin_amdgcn_wmma_f32_16x16x32_f16(false, a0, false, BiB,
                                                    (short)0, zero8, false, false);
#pragma unroll
    for (int r = 0; r < 8; ++r) { hA[r] += biA; hB[r] += biB; }

#pragma unroll
    for (int r = 0; r < 8; ++r) {
        ldsH[(mofs + r) * 32 + col]      = hA[r];
        ldsH[(mofs + r) * 32 + 16 + col] = hB[r];
    }
    __syncthreads();

    // ---- Sequential Euler integration, fused output projection ----
    for (int t = 0; t < T_STEPS; ++t) {
        // h_t in A layout (f16) from LDS
        v16h ah = load_a(ldsH, 32, 32, col, half);

        // y_t = h_t @ Wl + bl  -> out[b, t, :]
        v8f y = __builtin_amdgcn_wmma_f32_16x16x32_f16(false, ah, false, BL,
                                                       (short)0, zero8, false, false);
        if (col < OUTD) {
#pragma unroll
            for (int r = 0; r < 8; ++r) {
                size_t b = (size_t)(rowbase + mofs + r);
                out[b * (size_t)(T_STEPS * OUTD) + (size_t)t * OUTD + col] = y[r] + blv;
            }
        }
        if (t == T_STEPS - 1) break;

        const float dt = times[t + 1] - times[t];

        // layer 0: tanh(h @ Wf0 + bf0)
        v8f d = __builtin_amdgcn_wmma_f32_16x16x32_f16(false, ah, false, B0,
                                                       (short)0, zero8, false, false);
        __syncthreads();
#pragma unroll
        for (int r = 0; r < 8; ++r)
            ldsZ[(mofs + r) * 16 + col] = (col < HHD) ? tanhf(d[r] + b0v) : 0.0f;
        __syncthreads();
        v16h az = load_a(ldsZ, 16, HHD, col, half);

        // layer 1
        d = __builtin_amdgcn_wmma_f32_16x16x32_f16(false, az, false, B1,
                                                   (short)0, zero8, false, false);
        __syncthreads();
#pragma unroll
        for (int r = 0; r < 8; ++r)
            ldsZ[(mofs + r) * 16 + col] = (col < HHD) ? tanhf(d[r] + b1v) : 0.0f;
        __syncthreads();
        az = load_a(ldsZ, 16, HHD, col, half);

        // layer 2
        d = __builtin_amdgcn_wmma_f32_16x16x32_f16(false, az, false, B2,
                                                   (short)0, zero8, false, false);
        __syncthreads();
#pragma unroll
        for (int r = 0; r < 8; ++r)
            ldsZ[(mofs + r) * 16 + col] = (col < HHD) ? tanhf(d[r] + b2v) : 0.0f;
        __syncthreads();
        az = load_a(ldsZ, 16, HHD, col, half);

        // layer 3 (HH -> HID, two N-halves) + Euler update in D layout
        v8f d3a = __builtin_amdgcn_wmma_f32_16x16x32_f16(false, az, false, B3A,
                                                         (short)0, zero8, false, false);
        v8f d3b = __builtin_amdgcn_wmma_f32_16x16x32_f16(false, az, false, B3B,
                                                         (short)0, zero8, false, false);
#pragma unroll
        for (int r = 0; r < 8; ++r) {
            hA[r] += dt * (d3a[r] + b3A);
            hB[r] += dt * (d3b[r] + b3B);
        }

        __syncthreads();
#pragma unroll
        for (int r = 0; r < 8; ++r) {
            ldsH[(mofs + r) * 32 + col]      = hA[r];
            ldsH[(mofs + r) * 32 + 16 + col] = hB[r];
        }
        __syncthreads();
    }
}

extern "C" void kernel_launch(void* const* d_in, const int* in_sizes, int n_in,
                              void* d_out, int out_size, void* d_ws, size_t ws_size,
                              hipStream_t stream) {
    (void)in_sizes; (void)n_in; (void)out_size; (void)d_ws; (void)ws_size;
    const float* times = (const float*)d_in[0];
    const float* initial = (const float*)d_in[1];
    const float* Wi  = (const float*)d_in[2];
    const float* bi  = (const float*)d_in[3];
    const float* Wf0 = (const float*)d_in[4];
    const float* bf0 = (const float*)d_in[5];
    const float* Wf1 = (const float*)d_in[6];
    const float* bf1 = (const float*)d_in[7];
    const float* Wf2 = (const float*)d_in[8];
    const float* bf2 = (const float*)d_in[9];
    const float* Wf3 = (const float*)d_in[10];
    const float* bf3 = (const float*)d_in[11];
    const float* Wl  = (const float*)d_in[12];
    const float* bl  = (const float*)d_in[13];
    float* out = (float*)d_out;

    dim3 grid(BATCH / 16);   // 256 independent wave-tiles
    dim3 block(32);          // one wave32 per block
    node_fused_kernel<<<grid, block, 0, stream>>>(times, initial, Wi, bi,
                                                  Wf0, bf0, Wf1, bf1,
                                                  Wf2, bf2, Wf3, bf3,
                                                  Wl, bl, out);
}
